// MultiScaleAttention_62912680952672
// MI455X (gfx1250) — matrix operand order repaired and checked by reading, not compiled, not run on hardware
//
#include <hip/hip_runtime.h>
#include <hip/hip_bf16.h>
#include <cstdint>
#include <cstddef>

// ---------- types ----------
typedef __attribute__((ext_vector_type(16))) __bf16       bf16x16;
typedef __attribute__((ext_vector_type(8)))  float        f32x8;
typedef __attribute__((ext_vector_type(4)))  unsigned int u32x4;

union FragAB { bf16x16 v; u32x4 u[2]; };

__device__ __forceinline__ __bf16 f2bf(float f) {
  union { float f; unsigned u; } in; in.f = f;
  unsigned r = in.u + 0x7FFFu + ((in.u >> 16) & 1u);   // round-to-nearest-even
  unsigned short h = (unsigned short)(r >> 16);
  union { unsigned short s; __bf16 b; } out; out.s = h;
  return out.b;
}

__device__ __forceinline__ f32x8 wmma_bf16(const FragAB& a, const FragAB& b, f32x8 c) {
  return __builtin_amdgcn_wmma_f32_16x16x32_bf16(false, a.v, false, b.v,
                                                 (short)0, c, false, false);
}

// 16B global -> LDS async copy (per lane), tracked by ASYNCcnt
__device__ __forceinline__ void async_cp16(const __bf16* gp, const __bf16* lp) {
  unsigned       lds = (unsigned)(unsigned long long)(const void*)lp; // LDS offset = addr[31:0]
  unsigned long long ga = (unsigned long long)gp;
  asm volatile("global_load_async_to_lds_b128 %0, %1, off"
               :: "v"(lds), "v"(ga) : "memory");
}
__device__ __forceinline__ void wait_async0() {
  asm volatile("s_wait_asynccnt 0x0" ::: "memory");
}

// ---------- problem constants ----------
#define C_DIM   512
#define NHEAD   16
#define HD      32
#define NR      49        // tokens per window
#define NRP     64        // padded
#define NWIN    2048      // B_ = 32*64
#define WH      32768     // windows*heads
#define MROWS   100352    // B*N
#define BM      128
#define BN      128
#define BK      32
#define KSTEPS  (C_DIM / BK)

// =====================================================================
// Kernel 1: qkv = x @ W_qkv^T  (NT bf16 WMMA GEMM, double-buffered LDS),
// epilogue scatters q (scaled) / k / v^T (padded) as bf16
// =====================================================================
__global__ __launch_bounds__(256) void qkv_gemm(
    const float* __restrict__ X, const float* __restrict__ Wqkv,
    __bf16* __restrict__ qg, __bf16* __restrict__ kg, __bf16* __restrict__ vT)
{
  __shared__ __align__(16) __bf16 As[2][BM * BK];
  __shared__ __align__(16) __bf16 Bs[2][BN * BK];

  const int m0   = blockIdx.x * BM;
  const int n0   = blockIdx.y * BN;
  const int tid  = threadIdx.x;
  const int lane = tid & 31;
  const int w    = tid >> 5;
  const int mw   = w & 3;          // 4 waves along M
  const int nw   = w >> 2;         // 2 waves along N
  const int lm   = lane & 15;
  const int kh   = lane >> 4;
  const int r0   = tid >> 3;       // 0..31
  const int c0   = (tid & 7) * 4;  // 0,4,..,28

  f32x8 acc[2][4];
#pragma unroll
  for (int mi = 0; mi < 2; ++mi)
#pragma unroll
    for (int nj = 0; nj < 4; ++nj) acc[mi][nj] = f32x8{};

  float4 ra[4], rb[4];
  auto loadTile = [&](int k0) {
#pragma unroll
    for (int i = 0; i < 4; ++i) {
      const int row = r0 + 32 * i;
      ra[i] = *(const float4*)(X    + (size_t)(m0 + row) * C_DIM + k0 + c0);
      rb[i] = *(const float4*)(Wqkv + (size_t)(n0 + row) * C_DIM + k0 + c0);
    }
  };
  auto storeTile = [&](int buf) {
#pragma unroll
    for (int i = 0; i < 4; ++i) {
      const int row = r0 + 32 * i;
      As[buf][row * BK + c0 + 0] = f2bf(ra[i].x); As[buf][row * BK + c0 + 1] = f2bf(ra[i].y);
      As[buf][row * BK + c0 + 2] = f2bf(ra[i].z); As[buf][row * BK + c0 + 3] = f2bf(ra[i].w);
      Bs[buf][row * BK + c0 + 0] = f2bf(rb[i].x); Bs[buf][row * BK + c0 + 1] = f2bf(rb[i].y);
      Bs[buf][row * BK + c0 + 2] = f2bf(rb[i].z); Bs[buf][row * BK + c0 + 3] = f2bf(rb[i].w);
    }
  };

  loadTile(0);
  storeTile(0);
  __syncthreads();

  for (int step = 0; step < KSTEPS; ++step) {
    const int cur = step & 1;
    if (step + 1 < KSTEPS) loadTile((step + 1) * BK);   // prefetch into regs

    FragAB af[2], bfr[4];
#pragma unroll
    for (int mi = 0; mi < 2; ++mi) {
      const int m = mw * 32 + mi * 16 + lm;
      af[mi].u[0] = *(const u32x4*)&As[cur][m * BK + kh * 8];
      af[mi].u[1] = *(const u32x4*)&As[cur][m * BK + 16 + kh * 8];
    }
#pragma unroll
    for (int nj = 0; nj < 4; ++nj) {
      const int n = nw * 64 + nj * 16 + lm;
      bfr[nj].u[0] = *(const u32x4*)&Bs[cur][n * BK + kh * 8];
      bfr[nj].u[1] = *(const u32x4*)&Bs[cur][n * BK + 16 + kh * 8];
    }
#pragma unroll
    for (int mi = 0; mi < 2; ++mi)
#pragma unroll
      for (int nj = 0; nj < 4; ++nj)
        acc[mi][nj] = wmma_bf16(af[mi], bfr[nj], acc[mi][nj]);

    if (step + 1 < KSTEPS) storeTile(cur ^ 1);
    __syncthreads();
  }

  const float scale = 0.17677669529663687f;   // 1/sqrt(32)
#pragma unroll
  for (int mi = 0; mi < 2; ++mi)
#pragma unroll
    for (int nj = 0; nj < 4; ++nj)
#pragma unroll
      for (int vr = 0; vr < 8; ++vr) {
        const int m = m0 + mw * 32 + mi * 16 + vr + 8 * kh;
        const int n = n0 + nw * 64 + nj * 16 + lm;
        const float val = acc[mi][nj][vr];
        const int s  = n >> 9;          // qkv selector
        const int c  = n & 511;
        const int hh = c >> 5;
        const int d  = c & 31;
        const int b_ = m / NR;
        const int r  = m - b_ * NR;
        const size_t widx = (size_t)b_ * NHEAD + hh;
        if (s == 0)      qg[(widx * NR + r) * HD + d] = f2bf(val * scale);
        else if (s == 1) kg[(widx * NR + r) * HD + d] = f2bf(val);
        else             vT[(widx * HD + d) * NRP + r] = f2bf(val);
      }
}

// =====================================================================
// Kernel 2: per-(window,head) attention: S=qk^T+bias, softmax, O=Pv
// =====================================================================
__global__ __launch_bounds__(128) void win_attn(
    const __bf16* __restrict__ qg, const __bf16* __restrict__ kg,
    const __bf16* __restrict__ vT, const float* __restrict__ bias_table,
    __bf16* __restrict__ Og, float* __restrict__ sums)
{
  __shared__ __align__(16) float  Sbuf[NRP * NRP];
  __shared__ __align__(16) __bf16 Pbuf[NRP * NRP];

  const int wh   = blockIdx.x;
  const int b_   = wh >> 4;
  const int h    = wh & 15;
  const int tid  = threadIdx.x;
  const int lane = tid & 31;
  const int mi   = tid >> 5;       // wave id == M tile
  const int lm   = lane & 15;
  const int kh   = lane >> 4;

  // ---- stage 1: S = q @ k^T (K = hd = 32, one WMMA per tile) ----
  {
    FragAB qa;
    const int m = mi * 16 + lm;
    if (m < NR) {
      const __bf16* p = qg + ((size_t)wh * NR + m) * HD;
      qa.u[0] = *(const u32x4*)(p + kh * 8);
      qa.u[1] = *(const u32x4*)(p + 16 + kh * 8);
    } else { qa.u[0] = u32x4{}; qa.u[1] = u32x4{}; }

    f32x8 acc[4];
#pragma unroll
    for (int nj = 0; nj < 4; ++nj) {
      FragAB kb;
      const int t = nj * 16 + lm;
      if (t < NR) {
        const __bf16* p = kg + ((size_t)wh * NR + t) * HD + kh * 8;
        kb.u[0] = *(const u32x4*)(p);
        kb.u[1] = *(const u32x4*)(p + 16);
      } else { kb.u[0] = u32x4{}; kb.u[1] = u32x4{}; }
      acc[nj] = wmma_bf16(qa, kb, f32x8{});
    }
#pragma unroll
    for (int nj = 0; nj < 4; ++nj)
#pragma unroll
      for (int vr = 0; vr < 8; ++vr)
        Sbuf[(mi * 16 + vr + 8 * kh) * NRP + nj * 16 + lm] = acc[nj][vr];
  }
  __syncthreads();

  // ---- stage 2: +bias, softmax (fp32), P -> bf16 ----
  if (tid < NR) {
    const int qi = tid / 7, qj = tid - qi * 7;
    float mx = -1e30f;
    for (int t = 0; t < NR; ++t) {
      const int ki = t / 7, kj = t - ki * 7;
      const int rel = (qi - ki + 6) * 13 + (qj - kj + 6);
      const float s = Sbuf[tid * NRP + t] + bias_table[rel * NHEAD + h];
      Sbuf[tid * NRP + t] = s;
      mx = fmaxf(mx, s);
    }
    float ssum = 0.f;
    for (int t = 0; t < NR; ++t) {
      const float e = __expf(Sbuf[tid * NRP + t] - mx);
      Sbuf[tid * NRP + t] = e;
      ssum += e;
    }
    const float inv = 1.f / ssum;
    for (int t = 0; t < NR; ++t) Pbuf[tid * NRP + t] = f2bf(Sbuf[tid * NRP + t] * inv);
    for (int t = NR; t < NRP; ++t) Pbuf[tid * NRP + t] = f2bf(0.f);
  } else if (tid < NRP) {
    for (int t = 0; t < NRP; ++t) Pbuf[tid * NRP + t] = f2bf(0.f);
  }
  __syncthreads();

  // ---- stage 3: O = P @ v (K = 64 padded, vT pad is zeroed) ----
  f32x8 oacc[2] = { f32x8{}, f32x8{} };
#pragma unroll
  for (int ks = 0; ks < 2; ++ks) {
    FragAB pa;
    const int m = mi * 16 + lm;
    pa.u[0] = *(const u32x4*)&Pbuf[m * NRP + ks * 32 + kh * 8];
    pa.u[1] = *(const u32x4*)&Pbuf[m * NRP + ks * 32 + 16 + kh * 8];
#pragma unroll
    for (int nj = 0; nj < 2; ++nj) {
      FragAB vb;
      const int d = nj * 16 + lm;
      const __bf16* p = vT + ((size_t)wh * HD + d) * NRP + ks * 32 + kh * 8;
      vb.u[0] = *(const u32x4*)(p);
      vb.u[1] = *(const u32x4*)(p + 16);
      oacc[nj] = wmma_bf16(pa, vb, oacc[nj]);
    }
  }

  // ---- epilogue: store bf16 O + SE partial sums ----
  float lsum = 0.f;
#pragma unroll
  for (int nj = 0; nj < 2; ++nj)
#pragma unroll
    for (int vr = 0; vr < 8; ++vr) {
      const int m = mi * 16 + vr + 8 * kh;
      if (m < NR) {
        const float val = oacc[nj][vr];
        Og[((size_t)b_ * NR + m) * C_DIM + h * HD + nj * 16 + lm] = f2bf(val);
        lsum += val;
      }
    }
  for (int off = 16; off; off >>= 1) lsum += __shfl_xor(lsum, off, 32);
  if (lane == 0) atomicAdd(&sums[b_], lsum);
}

// =====================================================================
// Kernel 3: window SE gate: y = sigmoid(relu(mean @ w1^T) @ w2^T)
// =====================================================================
__global__ __launch_bounds__(64) void se_kernel(
    const float* __restrict__ sums, const float* __restrict__ w1,
    const float* __restrict__ w2, float* __restrict__ y)
{
  __shared__ float mv[64];
  __shared__ float hid[16];
  const int b = blockIdx.x, t = threadIdx.x;
  mv[t] = sums[b * 64 + t] * (1.0f / (49.0f * 512.0f));
  __syncthreads();
  if (t < 16) {
    float a = 0.f;
    for (int g = 0; g < 64; ++g) a += mv[g] * w1[t * 64 + g];
    hid[t] = fmaxf(a, 0.f);
  }
  __syncthreads();
  float a = 0.f;
  for (int i = 0; i < 16; ++i) a += hid[i] * w2[t * 16 + i];
  y[b * 64 + t] = 1.f / (1.f + __expf(-a));
}

// =====================================================================
// Kernel 4: out = (O * y_gate) @ W_proj^T + b_proj
// A tile (bf16) is staged via GLOBAL_LOAD_ASYNC_TO_LDS_B128 (ASYNCcnt),
// B tile (fp32->bf16) via register-staged prefetch; double-buffered LDS.
// =====================================================================
__global__ __launch_bounds__(256) void proj_gemm(
    const __bf16* __restrict__ Og, const float* __restrict__ Wp,
    const float* __restrict__ bproj, const float* __restrict__ y,
    float* __restrict__ out)
{
  __shared__ __align__(16) __bf16 As[2][BM * BK];
  __shared__ __align__(16) __bf16 Bs[2][BN * BK];

  const int m0   = blockIdx.x * BM;
  const int n0   = blockIdx.y * BN;
  const int tid  = threadIdx.x;
  const int lane = tid & 31;
  const int w    = tid >> 5;
  const int mw   = w & 3;
  const int nw   = w >> 2;
  const int lm   = lane & 15;
  const int kh   = lane >> 4;
  const int r0   = tid >> 3;
  const int c0   = (tid & 7) * 4;

  f32x8 acc[2][4];
#pragma unroll
  for (int mi = 0; mi < 2; ++mi)
#pragma unroll
    for (int nj = 0; nj < 4; ++nj) acc[mi][nj] = f32x8{};

  // async A-tile: 512 chunks of 16B; each of 256 threads issues 2
  auto issueA = [&](int k0, int buf) {
#pragma unroll
    for (int i = 0; i < 2; ++i) {
      const int ch  = tid + i * 256;
      const int row = ch >> 2;
      const int q   = (ch & 3) * 8;
      async_cp16(Og + (size_t)(m0 + row) * C_DIM + k0 + q,
                 &As[buf][row * BK + q]);
    }
  };

  float4 rb[4];
  auto loadB = [&](int k0) {
#pragma unroll
    for (int i = 0; i < 4; ++i) {
      const int row = r0 + 32 * i;
      rb[i] = *(const float4*)(Wp + (size_t)(n0 + row) * C_DIM + k0 + c0);
    }
  };
  auto storeB = [&](int buf) {
#pragma unroll
    for (int i = 0; i < 4; ++i) {
      const int row = r0 + 32 * i;
      Bs[buf][row * BK + c0 + 0] = f2bf(rb[i].x); Bs[buf][row * BK + c0 + 1] = f2bf(rb[i].y);
      Bs[buf][row * BK + c0 + 2] = f2bf(rb[i].z); Bs[buf][row * BK + c0 + 3] = f2bf(rb[i].w);
    }
  };

  issueA(0, 0);
  loadB(0);
  storeB(0);
  wait_async0();
  __syncthreads();

  for (int step = 0; step < KSTEPS; ++step) {
    const int cur = step & 1;
    if (step + 1 < KSTEPS) {
      issueA((step + 1) * BK, cur ^ 1);   // async prefetch next A tile
      loadB((step + 1) * BK);             // reg prefetch next B tile
    }

    FragAB af[2], bfr[4];
#pragma unroll
    for (int mi = 0; mi < 2; ++mi) {
      const int m = mw * 32 + mi * 16 + lm;
      af[mi].u[0] = *(const u32x4*)&As[cur][m * BK + kh * 8];
      af[mi].u[1] = *(const u32x4*)&As[cur][m * BK + 16 + kh * 8];
    }
#pragma unroll
    for (int nj = 0; nj < 4; ++nj) {
      const int n = nw * 64 + nj * 16 + lm;
      bfr[nj].u[0] = *(const u32x4*)&Bs[cur][n * BK + kh * 8];
      bfr[nj].u[1] = *(const u32x4*)&Bs[cur][n * BK + 16 + kh * 8];
    }
#pragma unroll
    for (int mi = 0; mi < 2; ++mi)
#pragma unroll
      for (int nj = 0; nj < 4; ++nj)
        acc[mi][nj] = wmma_bf16(af[mi], bfr[nj], acc[mi][nj]);

    if (step + 1 < KSTEPS) {
      storeB(cur ^ 1);
      wait_async0();                      // next A tile landed in LDS
    }
    __syncthreads();
  }

#pragma unroll
  for (int mi = 0; mi < 2; ++mi)
#pragma unroll
    for (int nj = 0; nj < 4; ++nj)
#pragma unroll
      for (int vr = 0; vr < 8; ++vr) {
        const int m = m0 + mw * 32 + mi * 16 + vr + 8 * kh;
        const int n = n0 + nw * 64 + nj * 16 + lm;
        const int b_ = m / NR;
        out[(size_t)m * C_DIM + n] = y[b_] * acc[mi][nj][vr] + bproj[n];
      }
}

// =====================================================================
extern "C" void kernel_launch(void* const* d_in, const int* in_sizes, int n_in,
                              void* d_out, int out_size, void* d_ws, size_t ws_size,
                              hipStream_t stream) {
  const float* X    = (const float*)d_in[0];
  const float* Wqkv = (const float*)d_in[1];
  const float* bias = (const float*)d_in[2];
  const float* Wp   = (const float*)d_in[3];
  const float* bp   = (const float*)d_in[4];
  const float* w1   = (const float*)d_in[5];
  const float* w2   = (const float*)d_in[6];
  (void)in_sizes; (void)n_in; (void)out_size; (void)ws_size;

  char* ws = (char*)d_ws;
  const size_t SQ = (size_t)MROWS * C_DIM;      // 100352*512 elems
  const size_t VT = (size_t)WH * HD * NRP;      // padded v^T elems
  __bf16* qg   = (__bf16*)(ws);
  __bf16* kg   = (__bf16*)(ws + SQ * 2);
  __bf16* vT   = (__bf16*)(ws + 2 * SQ * 2);
  __bf16* Og   = (__bf16*)(ws + 2 * SQ * 2 + VT * 2);
  float*  sums = (float*) (ws + 3 * SQ * 2 + VT * 2);
  float*  yv   = (float*) (ws + 3 * SQ * 2 + VT * 2 + NWIN * 4);

  // zero v^T pad (NaN safety for padded K) and SE accumulators
  hipMemsetAsync(vT, 0, VT * 2, stream);
  hipMemsetAsync(sums, 0, NWIN * 4, stream);

  qkv_gemm <<<dim3(MROWS / BM, (3 * C_DIM) / BN), 256, 0, stream>>>(X, Wqkv, qg, kg, vT);
  win_attn <<<dim3(WH), 128, 0, stream>>>(qg, kg, vT, bias, Og, sums);
  se_kernel<<<dim3(32), 64, 0, stream>>>(sums, w1, w2, yv);
  proj_gemm<<<dim3(MROWS / BM, C_DIM / BN), 256, 0, stream>>>(Og, Wp, bp, yv, (float*)d_out);
}
